// GraphPoolingLayer_22935125360694
// MI455X (gfx1250) — compile-verified, matrix operand order checked by reading, not run on hardware
//
#include <hip/hip_runtime.h>

typedef __attribute__((ext_vector_type(16))) _Float16 v16h;
typedef __attribute__((ext_vector_type(8)))  float    v8f;
typedef __attribute__((ext_vector_type(4)))  float    v4f;

#define D_FEAT 128
#define TILE_M 16
#define WAVES_PER_BLOCK 8

// ---------------------------------------------------------------------------
// Zero the output accumulator (harness poisons d_out with 0xAA).
// ---------------------------------------------------------------------------
__global__ void zero_out_kernel(float* __restrict__ out, int n) {
    int i = (blockIdx.x * blockDim.x + threadIdx.x) * 4;
    if (i < n) {
        v4f z = {0.f, 0.f, 0.f, 0.f};
        *(v4f*)(out + i) = z;
    }
}

// ---------------------------------------------------------------------------
// Fused linear + row-softmax + sorted segment-sum.
// One wave32 computes one 16-row tile of Y = softmax(X*W + b) via
// v_wmma_f32_16x16x32_f16, then run-length-combines sorted segment ids and
// emits global FP32 atomics into the per-molecule output.
// ---------------------------------------------------------------------------
__global__ __launch_bounds__(256)
void graph_pool_wmma_kernel(const float* __restrict__ X,
                            const int*   __restrict__ seg,
                            const float* __restrict__ W,
                            const float* __restrict__ bias,
                            float*       __restrict__ out,
                            int n_atoms) {
    // W pre-swizzled into B-fragment layout: 4 k-steps * 8 n-tiles * 32 lanes
    // * 16 halves = 32 KB of LDS.
    __shared__ __align__(32) _Float16 ldsW[4 * 8 * 32 * 16];

    const int tid = threadIdx.x;

    // ---- Stage W (f32 -> f16) into per-lane fragment layout, once per block.
    // ISA 16-bit B layout (16x16x32): lane holds column n = lane&15;
    // half element j maps to K = kk*32 + 8*(lane>>4) + j        (j < 8)
    //                         K = kk*32 + 8*(lane>>4) + 16 + (j-8) (j >= 8)
    for (int s = tid; s < 1024; s += 256) {
        const int lane  = s & 31;
        const int nt    = (s >> 5) & 7;
        const int kk    = s >> 8;
        const int n     = nt * 16 + (lane & 15);
        const int kbase = kk * 32 + (lane >> 4) * 8;
        __align__(32) _Float16 tmp[16];
#pragma unroll
        for (int j = 0; j < 8; ++j)
            tmp[j]     = (_Float16)W[(kbase + j) * D_FEAT + n];
#pragma unroll
        for (int j = 0; j < 8; ++j)
            tmp[8 + j] = (_Float16)W[(kbase + 16 + j) * D_FEAT + n];
        *(v16h*)&ldsW[s * 16] = *(v16h*)tmp;
    }
    __syncthreads();

    const int wave = tid >> 5;
    const int lane = tid & 31;
    const int tile = blockIdx.x * WAVES_PER_BLOCK + wave;
    if (tile * TILE_M >= n_atoms) return;   // wave-uniform exit, EXEC stays full

    const int row0 = tile * TILE_M;
    const int lh   = lane >> 4;   // 0 -> rows 0..7, 1 -> rows 8..15 (C layout)
    const int ln   = lane & 15;   // column within 16-wide n-tile / A row

    // ---- Accumulators, initialized with the bias (bias is per-column).
    v8f acc[8];
#pragma unroll
    for (int t = 0; t < 8; ++t) {
        const float bv = bias[t * 16 + ln];
#pragma unroll
        for (int r = 0; r < 8; ++r) acc[t][r] = bv;
    }

    // ---- GEMM: 4 k-steps of 32, 8 n-tiles of 16 -> 32 WMMAs.
    const float* xrow = X + (size_t)(row0 + ln) * D_FEAT;
#pragma unroll
    for (int kk = 0; kk < 4; ++kk) {
        const int kbase = kk * 32 + lh * 8;
        // A fragment: K-map is two contiguous 8-float runs per lane.
        v4f p0 = *(const v4f*)(xrow + kbase);
        v4f p1 = *(const v4f*)(xrow + kbase + 4);
        v4f p2 = *(const v4f*)(xrow + kbase + 16);
        v4f p3 = *(const v4f*)(xrow + kbase + 20);
        __align__(32) _Float16 ah[16];
#pragma unroll
        for (int j = 0; j < 4; ++j) {
            ah[j]      = (_Float16)p0[j];
            ah[4 + j]  = (_Float16)p1[j];
            ah[8 + j]  = (_Float16)p2[j];
            ah[12 + j] = (_Float16)p3[j];
        }
        v16h a = *(v16h*)ah;
#pragma unroll
        for (int nt = 0; nt < 8; ++nt) {
            v16h bm = *(const v16h*)&ldsW[((kk * 8 + nt) * 32 + lane) * 16];
            acc[nt] = __builtin_amdgcn_wmma_f32_16x16x32_f16(
                false, a, false, bm, (short)0, acc[nt], false, false);
        }
    }

    // ---- Row softmax. Row M = lh*8 + r lives in acc[*][r] across the 16
    // lanes of this half; xor-shuffles with masks 1,2,4,8 stay inside a half.
#pragma unroll
    for (int r = 0; r < 8; ++r) {
        float m = acc[0][r];
#pragma unroll
        for (int t = 1; t < 8; ++t) m = fmaxf(m, acc[t][r]);
#pragma unroll
        for (int off = 1; off <= 8; off <<= 1)
            m = fmaxf(m, __shfl_xor(m, off, 32));
        float s = 0.f;
#pragma unroll
        for (int t = 0; t < 8; ++t) {
            float e = __expf(acc[t][r] - m);
            acc[t][r] = e;
            s += e;
        }
#pragma unroll
        for (int off = 1; off <= 8; off <<= 1)
            s += __shfl_xor(s, off, 32);
        const float rinv = 1.0f / s;
#pragma unroll
        for (int t = 0; t < 8; ++t) acc[t][r] *= rinv;
    }

    // ---- Sorted segment-sum: run-length-combine the 8 rows this lane holds,
    // then one FP32 atomic per (segment, column). ids are uniform across each
    // 16-lane half, so the branches are half-uniform.
    int sid[8];
#pragma unroll
    for (int r = 0; r < 8; ++r) sid[r] = seg[row0 + lh * 8 + r];

    float run[8];
#pragma unroll
    for (int t = 0; t < 8; ++t) run[t] = acc[t][0];
    int cur = sid[0];
#pragma unroll
    for (int r = 1; r < 8; ++r) {
        if (sid[r] != cur) {
#pragma unroll
            for (int t = 0; t < 8; ++t) {
                unsafeAtomicAdd(&out[(size_t)cur * D_FEAT + t * 16 + ln], run[t]);
                run[t] = 0.f;
            }
            cur = sid[r];
        }
#pragma unroll
        for (int t = 0; t < 8; ++t) run[t] += acc[t][r];
    }
#pragma unroll
    for (int t = 0; t < 8; ++t)
        unsafeAtomicAdd(&out[(size_t)cur * D_FEAT + t * 16 + ln], run[t]);
}

// ---------------------------------------------------------------------------
extern "C" void kernel_launch(void* const* d_in, const int* in_sizes, int n_in,
                              void* d_out, int out_size, void* d_ws, size_t ws_size,
                              hipStream_t stream) {
    const float* X    = (const float*)d_in[0];
    const int*   seg  = (const int*)d_in[1];
    const float* W    = (const float*)d_in[2];
    const float* bias = (const float*)d_in[3];
    float*       out  = (float*)d_out;

    const int n_atoms = in_sizes[1];   // segment_ids element count

    // Zero the molecule accumulator first (same stream -> ordered).
    const int zthreads = 256;
    const int zblocks  = (out_size / 4 + zthreads - 1) / zthreads;
    zero_out_kernel<<<zblocks, zthreads, 0, stream>>>(out, out_size);

    const int tiles  = (n_atoms + TILE_M - 1) / TILE_M;
    const int blocks = (tiles + WAVES_PER_BLOCK - 1) / WAVES_PER_BLOCK;
    graph_pool_wmma_kernel<<<blocks, 256, 0, stream>>>(X, seg, W, bias, out, n_atoms);
}
